// AdaOctConv_14328010900130
// MI455X (gfx1250) — compile-verified, matrix operand order checked by reading, not run on hardware
//
#include <hip/hip_runtime.h>

// ---------------- WMMA vector types (gfx1250, wave32) ----------------
typedef __attribute__((ext_vector_type(16))) _Float16 v16h;
typedef __attribute__((ext_vector_type(8)))  float    v8f;
typedef __attribute__((ext_vector_type(8)))  _Float16 h8;
typedef __attribute__((ext_vector_type(4)))  int      v4i;

#define Bn    4
#define Cc    256
#define Hh    128
#define Wh    128
#define SLOPE 0.01f

#define COB  4   // cout 16-tiles per wave (block) -> 64 cout
#define POSW 2   // position 16-tiles per wave    -> 32 positions/wave, 256/block

// ---- CDNA5 async global->LDS copy path (guarded; falls back to VGPR staging) ----
#if defined(__has_builtin)
#if __has_builtin(__builtin_amdgcn_global_load_async_to_lds_b128) && \
    __has_builtin(__builtin_amdgcn_s_wait_asynccnt)
#define HAVE_ASYNC_LDS 1
#endif
#endif
#ifndef HAVE_ASYNC_LDS
#define HAVE_ASYNC_LDS 0
#endif

#if HAVE_ASYNC_LDS
// toolchain-probed signature: (v4i* gsrc, v4i* ldst, imm offset, imm cpol)
#define CP_ASYNC16(gsrc, ldst)                                             \
    __builtin_amdgcn_global_load_async_to_lds_b128(                        \
        (v4i*)(gsrc), (v4i*)(ldst), 0, 0)
#endif

// =====================================================================
// Weight repack: OIHW f32 -> [co][(kh*3+kw)*256 + ci] f16
// =====================================================================
__global__ void repack_w(const float* __restrict__ w, _Float16* __restrict__ wk) {
    int idx = blockIdx.x * 256 + threadIdx.x;          // 256*2304
    if (idx >= 256 * 2304) return;
    int co  = idx / 2304;
    int r   = idx % 2304;
    int khw = r >> 8;
    int ci  = r & 255;
    wk[idx] = (_Float16)w[(co * 256 + ci) * 9 + khw];
}

// =====================================================================
// kernel_predict: 3x3 reflect conv over the 3x3 style maps
// =====================================================================
__global__ void predict_spatial(const float* __restrict__ s, const float* __restrict__ sw,
                                const float* __restrict__ sb, float* __restrict__ wspat) {
    int idx = blockIdx.x * 256 + threadIdx.x;          // B*C*9 = 9216
    if (idx >= Bn * Cc * 9) return;
    int q  = idx % 3;
    int p  = (idx / 3) % 3;
    int co = (idx / 9) % Cc;
    int b  = idx / (9 * Cc);
    float acc = sb[co];
    for (int ci = 0; ci < Cc; ++ci) {
        const float* sp = s  + (b * Cc + ci) * 9;
        const float* wp = sw + (co * Cc + ci) * 9;
#pragma unroll
        for (int kh = 0; kh < 3; ++kh) {
            int r1 = p + kh - 1; r1 = r1 < 0 ? -r1 : (r1 > 2 ? 4 - r1 : r1);
#pragma unroll
            for (int kw = 0; kw < 3; ++kw) {
                int r2 = q + kw - 1; r2 = r2 < 0 ? -r2 : (r2 > 2 ? 4 - r2 : r2);
                acc += sp[r1 * 3 + r2] * wp[kh * 3 + kw];
            }
        }
    }
    wspat[(b * Cc + co) * 9 + p * 3 + q] = acc;
}

// pooled[b][c] = mean over 3x3
__global__ void pool_s(const float* __restrict__ s, float* __restrict__ pooled) {
    int idx = blockIdx.x * 256 + threadIdx.x;          // B*C = 1024
    if (idx >= Bn * Cc) return;
    const float* p = s + idx * 9;
    float a = 0.f;
#pragma unroll
    for (int i = 0; i < 9; ++i) a += p[i];
    pooled[idx] = a * (1.f / 9.f);
}

// w_point / bias matvecs
__global__ void point_bias(const float* __restrict__ pooled,
                           const float* __restrict__ pw, const float* __restrict__ pb,
                           const float* __restrict__ bw, const float* __restrict__ bb,
                           float* __restrict__ wpoint, float* __restrict__ biasv) {
    int idx = blockIdx.x * 256 + threadIdx.x;          // B*C = 1024
    if (idx >= Bn * Cc) return;
    int b = idx >> 8, o = idx & 255;
    float a1 = pb[o], a2 = bb[o];
    const float* pr = pooled + b * Cc;
    for (int i = 0; i < Cc; ++i) {
        float pv = pr[i];
        a1 += pv * pw[o * Cc + i];
        a2 += pv * bw[o * Cc + i];
    }
    wpoint[idx] = a1;
    biasv[idx]  = a2;
}

// =====================================================================
// Instance-norm statistics: one block per (b,c); LDS tree reduction
// =====================================================================
__global__ void in_stats(const float* __restrict__ x, int HW,
                         float* __restrict__ mean, float* __restrict__ rstd) {
    __shared__ float s1[256];
    __shared__ float s2[256];
    int bc = blockIdx.x;
    const float* p = x + (size_t)bc * HW;
    float s = 0.f, q = 0.f;
    for (int i = threadIdx.x; i < HW; i += 256) {
        float v = p[i];
        s += v; q += v * v;
    }
    s1[threadIdx.x] = s; s2[threadIdx.x] = q;
    __syncthreads();
    for (int st = 128; st > 0; st >>= 1) {
        if ((int)threadIdx.x < st) {
            s1[threadIdx.x] += s1[threadIdx.x + st];
            s2[threadIdx.x] += s2[threadIdx.x + st];
        }
        __syncthreads();
    }
    if (threadIdx.x == 0) {
        float m = s1[0] / (float)HW;
        float v = s2[0] / (float)HW - m * m;
        mean[bc] = m;
        rstd[bc] = rsqrtf(v + 1e-5f);
    }
}

// =====================================================================
// Fused: normalize + per-sample depthwise 3x3 (reflect) + scale/bias
// writes NHWC f16 for the WMMA GEMM stage
// =====================================================================
__global__ void dw_norm_conv(const float* __restrict__ x, const float* __restrict__ mean,
                             const float* __restrict__ rstd, const float* __restrict__ wsp,
                             const float* __restrict__ wpoint, const float* __restrict__ biasv,
                             int H, int W, _Float16* __restrict__ y) {
    int idx = blockIdx.x * 256 + threadIdx.x;          // B*C*H*W
    int w = idx % W;
    int t = idx / W;
    int h = t % H; t /= H;
    int c = t % Cc;
    int b = t / Cc;
    int bc = b * Cc + c;
    float mu = mean[bc], rs = rstd[bc];
    const float* xp = x + (size_t)bc * H * W;
    const float* wk = wsp + bc * 9;
    float acc = 0.f;
#pragma unroll
    for (int kh = 0; kh < 3; ++kh) {
        int hh = h + kh - 1; hh = hh < 0 ? -hh : (hh >= H ? 2 * H - 2 - hh : hh);
#pragma unroll
        for (int kw = 0; kw < 3; ++kw) {
            int ww = w + kw - 1; ww = ww < 0 ? -ww : (ww >= W ? 2 * W - 2 - ww : ww);
            acc += (xp[hh * W + ww] - mu) * rs * wk[kh * 3 + kw];
        }
    }
    float val = acc * wpoint[bc] + biasv[bc];
    y[((size_t)((b * H + h) * W + w) << 8) + c] = (_Float16)val;
}

// =====================================================================
// avgpool2 on NHWC f16 (hf -> half res), c fastest
// =====================================================================
__global__ void avgpool2_nhwc(const _Float16* __restrict__ x, _Float16* __restrict__ y) {
    int idx = blockIdx.x * 256 + threadIdx.x;          // B*64*64*256
    int c  = idx & 255;
    int t  = idx >> 8;
    int wl = t & 63; t >>= 6;
    int hl = t & 63;
    int b  = t >> 6;
    size_t base = ((size_t)((b * Hh + hl * 2) * Wh + wl * 2) << 8) + c;
    float a = (float)x[base] + (float)x[base + 256]
            + (float)x[base + (size_t)Wh * 256] + (float)x[base + (size_t)Wh * 256 + 256];
    y[(size_t)idx] = (_Float16)(a * 0.25f);
}

// =====================================================================
// Implicit-GEMM 3x3 reflect conv via V_WMMA_F32_16X16X32_F16 with
//   - 4x2 register blocking per wave (64 cout x 32 positions, 8 accums)
//   - block-shared weight slab in LDS (64co x 256ci f16 = 32 KB),
//     double-buffered (64 KB of the 320 KB WGP LDS)
//   - slab refill via GLOBAL_LOAD_ASYNC_TO_LDS_B128 (ASYNCcnt) when the
//     toolchain exposes it: no VGPR round-trip, one barrier per slab
//   - optional second fused (input,weight) pair (OctConv cross-frequency)
// Weight layout: wk[co][(kh*3+kw)*256 + ci] f16. Input: NHWC f16.
// up=1: logical input is nearest-upsample2 of a half-res map.
// =====================================================================
__global__ __launch_bounds__(256) void conv3x3_wmma(
        const _Float16* __restrict__ xa, int upa, const _Float16* __restrict__ wa,
        const _Float16* __restrict__ xb, int upb, const _Float16* __restrict__ wb,
        const float* __restrict__ bias, int has_bias,
        int H, int W,
        _Float16* __restrict__ out_nhwc, float* __restrict__ out_nchw, int use_nchw) {
    __shared__ alignas(16) _Float16 slab[2][64 * 256];   // 64 KB double-buffered A

    const int tid  = threadIdx.x;
    const int lane = tid & 31;
    const int wave = tid >> 5;
    const int half = lane >> 4;                        // K-half selector
    const int mrow = lane & 15;                        // A row / B column

    const int posTilesPerRow = W >> 4;
    const int posBlocks = (Bn * H * posTilesPerRow) >> 4;   // 16 pos tiles / block
    const int cb   = blockIdx.x / posBlocks;                // cout block (0..3)
    const int pblk = blockIdx.x % posBlocks;

    // this wave's two position tiles
    int h_[POSW], b_[POSW], w0_[POSW];
#pragma unroll
    for (int p = 0; p < POSW; ++p) {
        int pt  = pblk * 16 + wave * POSW + p;
        w0_[p]  = (pt % posTilesPerRow) << 4;
        h_[p]   = (pt / posTilesPerRow) % H;
        b_[p]   = pt / (posTilesPerRow * H);
    }

    // cooperative slab fill mapping: thread -> 64 halfs of the weight slab
    const int co_l = tid >> 2;              // 0..63 local cout
    const int cq   = (tid & 3) << 6;        // 0,64,128,192 ci base

    v8f acc[COB][POSW] = {};

#pragma unroll 1
    for (int src = 0; src < 2; ++src) {
        const _Float16* X  = src ? xb : xa;
        const _Float16* Wt = src ? wb : wa;
        const int up       = src ? upb : upa;
        if (X == nullptr) continue;

        const _Float16* wsrc = Wt + (size_t)(cb * 64 + co_l) * 2304 + cq;
        _Float16* lbase = &slab[0][co_l * 256 + cq];
        _Float16* lalt  = &slab[1][co_l * 256 + cq];

        // prologue: fill slab 0 (khw = 0)
#if HAVE_ASYNC_LDS
        {
#pragma unroll
            for (int j = 0; j < 8; ++j)
                CP_ASYNC16(wsrc + j * 8, lbase + j * 8);
            __builtin_amdgcn_s_wait_asynccnt(0);
        }
        __syncthreads();
#else
        {
            h8 r0[8];
#pragma unroll
            for (int j = 0; j < 8; ++j) r0[j] = *(const h8*)(wsrc + j * 8);
#pragma unroll
            for (int j = 0; j < 8; ++j) *(h8*)(lbase + j * 8) = r0[j];
        }
        __syncthreads();
#endif

#pragma unroll 1
        for (int khw = 0; khw < 9; ++khw) {
            const int cur = khw & 1;

#if HAVE_ASYNC_LDS
            // kick off async refill of the other buffer (safe: end-of-iter
            // barrier of khw-1 guarantees everyone is done reading it)
            if (khw < 8) {
                const _Float16* gsrc = wsrc + (khw + 1) * 256;
                _Float16* ldst = cur ? lbase : lalt;
#pragma unroll
                for (int j = 0; j < 8; ++j)
                    CP_ASYNC16(gsrc + j * 8, ldst + j * 8);
            }
#else
            h8 r[8];
            if (khw < 8) {
#pragma unroll
                for (int j = 0; j < 8; ++j)
                    r[j] = *(const h8*)(wsrc + (khw + 1) * 256 + j * 8);
            }
#endif

            // B row pointers for both position tiles (reflect / upsample)
            const int dh = khw / 3 - 1, dw = khw % 3 - 1;
            const _Float16* xrowp[POSW];
#pragma unroll
            for (int p = 0; p < POSW; ++p) {
                int hh = h_[p] + dh;
                hh = hh < 0 ? -hh : (hh >= H ? 2 * H - 2 - hh : hh);
                int ww = w0_[p] + mrow + dw;
                ww = ww < 0 ? -ww : (ww >= W ? 2 * W - 2 - ww : ww);
                if (up)
                    xrowp[p] = X + (((size_t)((b_[p] * (H >> 1) + (hh >> 1)) * (W >> 1) + (ww >> 1))) << 8) + half * 8;
                else
                    xrowp[p] = X + (((size_t)((b_[p] * H + hh) * W + ww)) << 8) + half * 8;
            }

            // 8 K-steps of 32 over ci; A from LDS, B from global
#pragma unroll
            for (int s = 0; s < 8; ++s) {
                const int ci0 = s * 32;
                v16h Bf[POSW];
#pragma unroll
                for (int p = 0; p < POSW; ++p) {
                    h8 blo = *(const h8*)(xrowp[p] + ci0);
                    h8 bhi = *(const h8*)(xrowp[p] + ci0 + 16);
                    Bf[p] = __builtin_shufflevector(blo, bhi,
                              0, 1, 2, 3, 4, 5, 6, 7, 8, 9, 10, 11, 12, 13, 14, 15);
                }
#pragma unroll
                for (int t = 0; t < COB; ++t) {
                    const _Float16* ap = &slab[cur][(t * 16 + mrow) * 256 + ci0 + half * 8];
                    h8 alo = *(const h8*)ap;
                    h8 ahi = *(const h8*)(ap + 16);
                    v16h Af = __builtin_shufflevector(alo, ahi,
                                0, 1, 2, 3, 4, 5, 6, 7, 8, 9, 10, 11, 12, 13, 14, 15);
#pragma unroll
                    for (int p = 0; p < POSW; ++p)
                        acc[t][p] = __builtin_amdgcn_wmma_f32_16x16x32_f16(
                                false, Af, false, Bf[p], (short)0, acc[t][p], false, false);
                }
            }

#if HAVE_ASYNC_LDS
            // my async writes have landed; barrier also separates all waves'
            // reads of slab[cur] from next iteration's refill of it
            if (khw < 8) __builtin_amdgcn_s_wait_asynccnt(0);
            __syncthreads();
#else
            if (khw < 8) {
                __syncthreads();           // everyone done reading slab[cur^1]
#pragma unroll
                for (int j = 0; j < 8; ++j)
                    *(h8*)((cur ? lbase : lalt) + j * 8) = r[j];
                __syncthreads();
            }
#endif
        }
#if !HAVE_ASYNC_LDS
        __syncthreads();                   // protect slab[0] before next src prologue
#endif
    }

    // Epilogue: D[row = v + 8*half][col = mrow] per (t, p) tile
#pragma unroll
    for (int t = 0; t < COB; ++t) {
#pragma unroll
        for (int p = 0; p < POSW; ++p) {
#pragma unroll
            for (int v = 0; v < 8; ++v) {
                int coOut = cb * 64 + t * 16 + v + (half << 3);
                float val = acc[t][p][v];
                if (has_bias) val += bias[coOut];
                val = val >= 0.f ? val : SLOPE * val;
                int wcol = w0_[p] + mrow;
                if (use_nchw)
                    out_nchw[(((size_t)(b_[p] * Cc + coOut) * H) + h_[p]) * W + (size_t)wcol] = val;
                else
                    out_nhwc[(((size_t)((b_[p] * H + h_[p]) * W + wcol)) << 8) + coOut] = (_Float16)val;
            }
        }
    }
}

// =====================================================================
// Host side
// =====================================================================
extern "C" void kernel_launch(void* const* d_in, const int* in_sizes, int n_in,
                              void* d_out, int out_size, void* d_ws, size_t ws_size,
                              hipStream_t stream) {
    (void)in_sizes; (void)n_in; (void)out_size; (void)ws_size;
    const float* c_hf = (const float*)d_in[0];
    const float* c_lf = (const float*)d_in[1];
    const float* s_hf = (const float*)d_in[2];
    const float* s_lf = (const float*)d_in[3];
    const float* h_sw = (const float*)d_in[4];
    const float* h_sb = (const float*)d_in[5];
    const float* h_pw = (const float*)d_in[6];
    const float* h_pb = (const float*)d_in[7];
    const float* h_bw = (const float*)d_in[8];
    const float* h_bb = (const float*)d_in[9];
    const float* l_sw = (const float*)d_in[10];
    const float* l_sb = (const float*)d_in[11];
    const float* l_pw = (const float*)d_in[12];
    const float* l_pb = (const float*)d_in[13];
    const float* l_bw = (const float*)d_in[14];
    const float* l_bb = (const float*)d_in[15];
    const float* ada_h_w = (const float*)d_in[16];
    const float* ada_h_b = (const float*)d_in[17];
    const float* ada_l_w = (const float*)d_in[18];
    const float* ada_l_b = (const float*)d_in[19];
    const float* w_h2h = (const float*)d_in[20];
    const float* w_l2h = (const float*)d_in[21];
    const float* w_h2l = (const float*)d_in[22];
    const float* w_l2l = (const float*)d_in[23];

    float* out_hf = (float*)d_out;                         // [4,256,128,128]
    float* out_lf = out_hf + (size_t)Bn * Cc * Hh * Wh;    // [4,256,64,64]

    // ---- workspace carve-up ----
    char* ws = (char*)d_ws;
    size_t off = 0;
    auto take = [&](size_t bytes) -> void* {
        void* p = ws + off;
        off += (bytes + 255) & ~(size_t)255;
        return p;
    };
    const size_t WKB = (size_t)Cc * 2304 * sizeof(_Float16);
    _Float16* wk_adah = (_Float16*)take(WKB);
    _Float16* wk_adal = (_Float16*)take(WKB);
    _Float16* wk_h2h  = (_Float16*)take(WKB);
    _Float16* wk_l2h  = (_Float16*)take(WKB);
    _Float16* wk_l2l  = (_Float16*)take(WKB);
    _Float16* wk_h2l  = (_Float16*)take(WKB);
    const size_t NHF = (size_t)Bn * Hh * Wh * Cc;
    const size_t NLF = NHF / 4;
    _Float16* y_h    = (_Float16*)take(NHF * 2);
    _Float16* y_l    = (_Float16*)take(NLF * 2);
    _Float16* hf16   = (_Float16*)take(NHF * 2);
    _Float16* lf16   = (_Float16*)take(NLF * 2);
    _Float16* hfpool = (_Float16*)take(NLF * 2);
    float* wspat_h  = (float*)take((size_t)Bn * Cc * 9 * 4);
    float* wspat_l  = (float*)take((size_t)Bn * Cc * 9 * 4);
    float* pooled_h = (float*)take((size_t)Bn * Cc * 4);
    float* pooled_l = (float*)take((size_t)Bn * Cc * 4);
    float* wpoint_h = (float*)take((size_t)Bn * Cc * 4);
    float* biasv_h  = (float*)take((size_t)Bn * Cc * 4);
    float* wpoint_l = (float*)take((size_t)Bn * Cc * 4);
    float* biasv_l  = (float*)take((size_t)Bn * Cc * 4);
    float* mean_h   = (float*)take((size_t)Bn * Cc * 4);
    float* rstd_h   = (float*)take((size_t)Bn * Cc * 4);
    float* mean_l   = (float*)take((size_t)Bn * Cc * 4);
    float* rstd_l   = (float*)take((size_t)Bn * Cc * 4);

    // ---- stage 0: repack the six big conv weights ----
    {
        int g = (256 * 2304 + 255) / 256;
        repack_w<<<g, 256, 0, stream>>>(ada_h_w, wk_adah);
        repack_w<<<g, 256, 0, stream>>>(ada_l_w, wk_adal);
        repack_w<<<g, 256, 0, stream>>>(w_h2h,  wk_h2h);
        repack_w<<<g, 256, 0, stream>>>(w_l2h,  wk_l2h);
        repack_w<<<g, 256, 0, stream>>>(w_l2l,  wk_l2l);
        repack_w<<<g, 256, 0, stream>>>(w_h2l,  wk_h2l);
    }

    // ---- stage 1: kernel_predict ----
    predict_spatial<<<36, 256, 0, stream>>>(s_hf, h_sw, h_sb, wspat_h);
    predict_spatial<<<36, 256, 0, stream>>>(s_lf, l_sw, l_sb, wspat_l);
    pool_s<<<4, 256, 0, stream>>>(s_hf, pooled_h);
    pool_s<<<4, 256, 0, stream>>>(s_lf, pooled_l);
    point_bias<<<4, 256, 0, stream>>>(pooled_h, h_pw, h_pb, h_bw, h_bb, wpoint_h, biasv_h);
    point_bias<<<4, 256, 0, stream>>>(pooled_l, l_pw, l_pb, l_bw, l_bb, wpoint_l, biasv_l);

    // ---- stage 2: instance-norm stats ----
    in_stats<<<Bn * Cc, 256, 0, stream>>>(c_hf, Hh * Wh, mean_h, rstd_h);
    in_stats<<<Bn * Cc, 256, 0, stream>>>(c_lf, (Hh / 2) * (Wh / 2), mean_l, rstd_l);

    // ---- stage 3: normalize + dynamic depthwise + scale/bias -> NHWC f16 ----
    dw_norm_conv<<<(int)(NHF / 256), 256, 0, stream>>>(
        c_hf, mean_h, rstd_h, wspat_h, wpoint_h, biasv_h, Hh, Wh, y_h);
    dw_norm_conv<<<(int)(NLF / 256), 256, 0, stream>>>(
        c_lf, mean_l, rstd_l, wspat_l, wpoint_l, biasv_l, Hh / 2, Wh / 2, y_l);

    // ---- stage 4: shared ada 3x3 convs (WMMA) + bias + LeakyReLU ----
    {
        int posBlocks_h = (Bn * Hh * (Wh / 16)) / 16;        // 256
        conv3x3_wmma<<<4 * posBlocks_h, 256, 0, stream>>>(
            y_h, 0, wk_adah, nullptr, 0, nullptr, ada_h_b, 1,
            Hh, Wh, hf16, nullptr, 0);
        int posBlocks_l = (Bn * (Hh / 2) * ((Wh / 2) / 16)) / 16;  // 64
        conv3x3_wmma<<<4 * posBlocks_l, 256, 0, stream>>>(
            y_l, 0, wk_adal, nullptr, 0, nullptr, ada_l_b, 1,
            Hh / 2, Wh / 2, lf16, nullptr, 0);
    }

    // ---- stage 5: avgpool2(hf) for the h2l branch ----
    avgpool2_nhwc<<<(int)(NLF / 256), 256, 0, stream>>>(hf16, hfpool);

    // ---- stage 6: OctConv fusion (dual-input WMMA GEMMs) -> d_out (NCHW f32) ----
    {
        int posBlocks_h = (Bn * Hh * (Wh / 16)) / 16;
        conv3x3_wmma<<<4 * posBlocks_h, 256, 0, stream>>>(
            hf16, 0, wk_h2h, lf16, 1, wk_l2h, nullptr, 0,
            Hh, Wh, nullptr, out_hf, 1);
        int posBlocks_l = (Bn * (Hh / 2) * ((Wh / 2) / 16)) / 16;
        conv3x3_wmma<<<4 * posBlocks_l, 256, 0, stream>>>(
            lf16, 0, wk_l2l, hfpool, 0, wk_h2l, nullptr, 0,
            Hh / 2, Wh / 2, nullptr, out_lf, 1);
    }
}
